// GIN_32478542692611
// MI455X (gfx1250) — matrix operand order repaired and checked by reading, not compiled, not run on hardware
//
#include <hip/hip_runtime.h>

// ---------------- problem constants ----------------
static constexpr int N_NODES = 250000;
static constexpr int N_EDGES = 500000;
static constexpr int D       = 128;
static constexpr int L_LAYERS= 4;
static constexpr int G_GRAPH = 10000;
static constexpr int OUT_DIM = 128;
static constexpr int AV      = 120;
static constexpr int BV      = 6;
static constexpr int NPG     = 25;       // nodes per graph (contiguous)
static constexpr float EPS_GN = 1e-6f;

typedef __attribute__((ext_vector_type(16))) _Float16 v16h;
typedef __attribute__((ext_vector_type(2)))  _Float16 h2;
typedef __attribute__((ext_vector_type(8)))  float    v8f;

// ---------------- zero fill ----------------
__global__ void zero_f4(float4* __restrict__ p) {
    int i = blockIdx.x * 256 + threadIdx.x;
    p[i] = make_float4(0.f, 0.f, 0.f, 0.f);
}

// ---------------- weight pack into WMMA-B f16 layout ----------------
// B-matrix (32x16 f16 per k-tile): lane L holds column n = L&15,
// k-range = kt*32 + ((L>=16)?16:0) + h, h=0..15 -> 16 contiguous halfs per lane.
__global__ void pack_w(const float* __restrict__ W, _Float16* __restrict__ out,
                       int K, int NCOLS) {
    int tid = blockIdx.x * 256 + threadIdx.x;
    int total = K * NCOLS;
    if (tid >= total) return;
    int NT   = NCOLS >> 4;
    int hh   = tid & 15;
    int lane = (tid >> 4) & 31;
    int ntk  = tid >> 9;
    int nt   = ntk % NT;
    int kt   = ntk / NT;
    int k = kt * 32 + ((lane >= 16) ? 16 : 0) + hh;
    int n = nt * 16 + (lane & 15);
    out[tid] = (_Float16)W[(size_t)k * NCOLS + n];
}

// ---------------- atom encoder: h[n,:] = sum_f atom_emb[f, nfeat[n,f], :] ----
__global__ void atom_enc(const int* __restrict__ nfeat,
                         const float* __restrict__ aemb,
                         float* __restrict__ h) {
    int gt = blockIdx.x * 256 + threadIdx.x;       // N*32 threads exactly
    int n  = gt >> 5;
    int d0 = (gt & 31) << 2;
    float4 acc = make_float4(0.f, 0.f, 0.f, 0.f);
#pragma unroll
    for (int f = 0; f < 9; ++f) {
        int idx = nfeat[n * 9 + f];
        float4 v = *(const float4*)(aemb + ((size_t)f * AV + idx) * D + d0);
        acc.x += v.x; acc.y += v.y; acc.z += v.z; acc.w += v.w;
    }
    *(float4*)(h + (size_t)n * D + d0) = acc;
}

// ---------------- edge message + scatter-add ----------------
__global__ void edge_msg(const int* __restrict__ src, const int* __restrict__ dst,
                         const int* __restrict__ efeat,
                         const float* __restrict__ bemb,   // [3][BV][D] for this layer
                         const float* __restrict__ h,
                         float* __restrict__ agg) {
    int gt = blockIdx.x * 256 + threadIdx.x;       // E*32 threads exactly
    int e  = gt >> 5;
    int d0 = (gt & 31) << 2;
    int s  = src[e];
    int t  = dst[e];
    float4 v = *(const float4*)(h + (size_t)s * D + d0);
#pragma unroll
    for (int f = 0; f < 3; ++f) {
        int idx = efeat[e * 3 + f];
        float4 be = *(const float4*)(bemb + ((size_t)f * BV + idx) * D + d0);
        v.x += be.x; v.y += be.y; v.z += be.z; v.w += be.w;
    }
    float* p = agg + (size_t)t * D + d0;
    unsafeAtomicAdd(p + 0, v.x);
    unsafeAtomicAdd(p + 1, v.y);
    unsafeAtomicAdd(p + 2, v.z);
    unsafeAtomicAdd(p + 3, v.w);
}

// ---------------- WMMA GEMM: Y[M,NCOLS] = (X1 (+X2)) @ Wpack + bias ----------
// block = 128 threads = 4 waves; block owns 16 rows x NCOLS cols.
// Wave w owns n-tiles [w*NTW, (w+1)*NTW).
template<int K, int NCOLS, int NTW, bool HASX2>
__global__ void gemm_wmma(const float* __restrict__ X1, const float* __restrict__ X2,
                          const _Float16* __restrict__ Wp,
                          const float* __restrict__ bias,
                          float* __restrict__ Y) {
    constexpr int NT = NCOLS / 16;
    __shared__ _Float16 As[16][K];

    const int row0 = blockIdx.x * 16;
    const int tid  = threadIdx.x;

    // stage A tile (f32 -> f16, fuse GIN self-term add)
    for (int i = tid; i < 16 * K; i += 128) {
        int r = i / K, c = i % K;
        float v = X1[(size_t)(row0 + r) * K + c];
        if (HASX2) v += X2[(size_t)(row0 + r) * K + c];
        As[r][c] = (_Float16)v;
    }
    __syncthreads();

    const int lane = tid & 31;
    const int wave = tid >> 5;
    const int m    = lane & 15;
    const int hi8  = (lane >> 4) << 3;   // 8 for upper half-wave, doubled to a +16 K-offset (A layout)

    v8f acc[NTW];
#pragma unroll
    for (int t = 0; t < NTW; ++t) acc[t] = (v8f){0.f,0.f,0.f,0.f,0.f,0.f,0.f,0.f};

#pragma unroll
    for (int kt = 0; kt < K / 32; ++kt) {
        // A 16x32 f16 per-lane layout: 16 contiguous K per lane (k = 16*half + s), matching B pack
        v16h a;
#pragma unroll
        for (int j = 0; j < 8; ++j) {
            int k0 = kt * 32 + (hi8 << 1) + (j << 1);
            h2 p = *(const h2*)&As[m][k0];
            a[2 * j]     = p.x;
            a[2 * j + 1] = p.y;
        }
#pragma unroll
        for (int t = 0; t < NTW; ++t) {
            int nt = wave * NTW + t;
            v16h b = *(const v16h*)(Wp + (((size_t)kt * NT + nt) * 32 + lane) * 16);
            acc[t] = __builtin_amdgcn_wmma_f32_16x16x32_f16(
                false, a, false, b, (short)0, acc[t], false, false);
        }
    }

    // store C (8 VGPRs: VGPR r -> M = r + (lane>=16 ? 8 : 0), N = lane&15)
    const int rplus = (lane >> 4) << 3;
#pragma unroll
    for (int t = 0; t < NTW; ++t) {
        int col = (wave * NTW + t) * 16 + (lane & 15);
        float bv = bias[col];
#pragma unroll
        for (int r = 0; r < 8; ++r) {
            Y[(size_t)(row0 + r + rplus) * NCOLS + col] = acc[t][r] + bv;
        }
    }
}

// ---------------- GraphNorm (per-graph = 25 contiguous rows), fused relu/residual
template<int C>
__global__ void graphnorm(const float* __restrict__ X,
                          const float* __restrict__ w, const float* __restrict__ b,
                          const float* __restrict__ ms,
                          const float* __restrict__ res,   // nullable residual
                          float* __restrict__ Y, int do_relu) {
    int g = blockIdx.x;
    int c = threadIdx.x;
    const float* base = X + (size_t)g * NPG * C + c;
    float x[NPG];
    float s = 0.f;
#pragma unroll
    for (int r = 0; r < NPG; ++r) { x[r] = base[(size_t)r * C]; s += x[r]; }
    float mean = s * (1.0f / NPG);
    float msub = mean * ms[c];
    float v = 0.f;
#pragma unroll
    for (int r = 0; r < NPG; ++r) { x[r] -= msub; v += x[r] * x[r]; }
    float inv = rsqrtf(v * (1.0f / NPG) + EPS_GN);
    float wc = w[c], bc = b[c];
    float* yb = Y + (size_t)g * NPG * C + c;
    const float* rb = res ? (res + (size_t)g * NPG * C + c) : nullptr;
#pragma unroll
    for (int r = 0; r < NPG; ++r) {
        float o = wc * x[r] * inv + bc;
        if (do_relu) o = fmaxf(o, 0.f);
        if (res) o += rb[(size_t)r * C];
        yb[(size_t)r * C] = o;
    }
}

// ---------------- mean pooling over each graph's 25 rows ----------------
__global__ void pool_mean(const float* __restrict__ h, float* __restrict__ g) {
    int gr = blockIdx.x;
    int c  = threadIdx.x;                  // D threads
    const float* base = h + (size_t)gr * NPG * D + c;
    float s = 0.f;
#pragma unroll
    for (int r = 0; r < NPG; ++r) s += base[(size_t)r * D];
    g[(size_t)gr * D + c] = s * (1.0f / NPG);
}

// ---------------- host-side launch ----------------
extern "C" void kernel_launch(void* const* d_in, const int* in_sizes, int n_in,
                              void* d_out, int out_size, void* d_ws, size_t ws_size,
                              hipStream_t stream) {
    const int*   nfeat    = (const int*)  d_in[0];
    const int*   efeat    = (const int*)  d_in[1];
    const int*   src      = (const int*)  d_in[2];
    const int*   dst      = (const int*)  d_in[3];
    // d_in[4] graph_ids: contiguous blocks of 25, handled structurally
    const float* atom_emb = (const float*)d_in[5];
    const float* bond_emb = (const float*)d_in[6];
    const float* W1       = (const float*)d_in[7];
    const float* b1       = (const float*)d_in[8];
    const float* gn1_w    = (const float*)d_in[9];
    const float* gn1_b    = (const float*)d_in[10];
    const float* gn1_ms   = (const float*)d_in[11];
    const float* W2       = (const float*)d_in[12];
    const float* b2       = (const float*)d_in[13];
    const float* gn2_w    = (const float*)d_in[14];
    const float* gn2_b    = (const float*)d_in[15];
    const float* gn2_ms   = (const float*)d_in[16];
    const float* Wp       = (const float*)d_in[17];
    const float* bp       = (const float*)d_in[18];
    float* out = (float*)d_out;

    // workspace carve-up (all offsets 256B-aligned)
    char* ws = (char*)d_ws;
    float* h    = (float*)ws; ws += (size_t)N_NODES * D * 4;           // 128 MB
    float* agg  = (float*)ws; ws += (size_t)N_NODES * D * 4;           // 128 MB (reused as t2)
    float* t1   = (float*)ws; ws += (size_t)N_NODES * 2 * D * 4;       // 256 MB
    float* gbuf = (float*)ws; ws += (size_t)G_GRAPH * D * 4;           // 5 MB
    _Float16* w1p = (_Float16*)ws; ws += (size_t)L_LAYERS * D * 2 * D * 2;
    _Float16* w2p = (_Float16*)ws; ws += (size_t)L_LAYERS * 2 * D * D * 2;
    _Float16* wpp = (_Float16*)ws; ws += (size_t)D * OUT_DIM * 2;

    // ---- pack weights into WMMA B layout (f16) ----
    for (int l = 0; l < L_LAYERS; ++l) {
        pack_w<<<(D * 2 * D + 255) / 256, 256, 0, stream>>>(
            W1 + (size_t)l * D * 2 * D, w1p + (size_t)l * D * 2 * D, D, 2 * D);
        pack_w<<<(2 * D * D + 255) / 256, 256, 0, stream>>>(
            W2 + (size_t)l * 2 * D * D, w2p + (size_t)l * 2 * D * D, 2 * D, D);
    }
    pack_w<<<(D * OUT_DIM + 255) / 256, 256, 0, stream>>>(Wp, wpp, D, OUT_DIM);

    // ---- atom encoder ----
    atom_enc<<<(N_NODES * 32) / 256, 256, 0, stream>>>(nfeat, atom_emb, h);

    // ---- conv layers ----
    for (int l = 0; l < L_LAYERS; ++l) {
        zero_f4<<<(N_NODES * D / 4) / 256, 256, 0, stream>>>((float4*)agg);

        edge_msg<<<(N_EDGES * 32) / 256, 256, 0, stream>>>(
            src, dst, efeat, bond_emb + (size_t)l * 3 * BV * D, h, agg);

        // rst = (agg + h) @ W1 + b1    [N, 256]
        gemm_wmma<128, 256, 4, true><<<N_NODES / 16, 128, 0, stream>>>(
            agg, h, w1p + (size_t)l * D * 2 * D, b1 + (size_t)l * 2 * D, t1);

        // GraphNorm1 + relu (in place)
        graphnorm<256><<<G_GRAPH, 256, 0, stream>>>(
            t1, gn1_w + (size_t)l * 2 * D, gn1_b + (size_t)l * 2 * D,
            gn1_ms + (size_t)l * 2 * D, nullptr, t1, 1);

        // t2 = relu(gn1) @ W2 + b2    [N, 128]  (t2 aliases agg)
        gemm_wmma<256, 128, 2, false><<<N_NODES / 16, 128, 0, stream>>>(
            t1, nullptr, w2p + (size_t)l * 2 * D * D, b2 + (size_t)l * D, agg);

        // GraphNorm2 (+relu if not last) + residual -> h (in place)
        graphnorm<128><<<G_GRAPH, 128, 0, stream>>>(
            agg, gn2_w + (size_t)l * D, gn2_b + (size_t)l * D,
            gn2_ms + (size_t)l * D, h, h, (l < L_LAYERS - 1) ? 1 : 0);
    }

    // ---- mean pool + final projection ----
    pool_mean<<<G_GRAPH, D, 0, stream>>>(h, gbuf);
    gemm_wmma<128, 128, 2, false><<<G_GRAPH / 16, 128, 0, stream>>>(
        gbuf, nullptr, wpp, bp, out);
}
